// StackedLstm_20469814132831
// MI455X (gfx1250) — matrix operand order repaired and checked by reading, not compiled, not run on hardware
//
#include <hip/hip_runtime.h>
#include <hip/hip_bf16.h>

typedef __bf16 bf16_t;
typedef __attribute__((ext_vector_type(16))) __bf16 v16bf;
typedef __attribute__((ext_vector_type(8)))  float  v8f;

#define HID   1024
#define G4    4096
#define BATCH 64
#define SEQT  512

// ---------------------------------------------------------------------------
// fp32 -> bf16 conversion (weights), grid-stride
// ---------------------------------------------------------------------------
__global__ __launch_bounds__(256)
void f32_to_bf16_kernel(const float* __restrict__ src, bf16_t* __restrict__ dst, int n) {
    for (int i = blockIdx.x * blockDim.x + threadIdx.x; i < n; i += gridDim.x * blockDim.x)
        dst[i] = (bf16_t)src[i];
}

// ---------------------------------------------------------------------------
// strided x_t gather + fp32->bf16: dst[b*HID+k] = x[(b*SEQT + t)*HID + k]
// src already offset by t*HID; row stride SEQT*HID.
// ---------------------------------------------------------------------------
__global__ __launch_bounds__(256)
void gather_x_bf16_kernel(const float* __restrict__ src, bf16_t* __restrict__ dst) {
    const int b = blockIdx.x;          // 64 blocks
    const int k0 = threadIdx.x * 4;    // 256 threads * 4 = 1024
    const float4 v = *(const float4*)(src + (size_t)b * (SEQT * HID) + k0);
    bf16_t* d = dst + (size_t)b * HID + k0;
    d[0] = (bf16_t)v.x; d[1] = (bf16_t)v.y; d[2] = (bf16_t)v.z; d[3] = (bf16_t)v.w;
}

// ---------------------------------------------------------------------------
// zero-init recurrent states (fp32 h,c and bf16 h copies)
// ---------------------------------------------------------------------------
__global__ __launch_bounds__(256)
void init_states_kernel(float* h0, float* c0, float* h1, float* c1,
                        bf16_t* hb0, bf16_t* hb1) {
    int i = blockIdx.x * blockDim.x + threadIdx.x;
    if (i < BATCH * HID) {
        h0[i] = 0.f; c0[i] = 0.f; h1[i] = 0.f; c1[i] = 0.f;
        hb0[i] = (bf16_t)0.f; hb1[i] = (bf16_t)0.f;
    }
}

// ---------------------------------------------------------------------------
// C[64 x 4096] = A(64 x 1024, bf16, row stride lda) * W(4096 x 1024, bf16)^T + bias
// One wave per 16x16 output tile; K-loop of v_wmma_f32_16x16x32_bf16.
// Grid: 256 blocks x 128 threads = 1024 waves = (64/16)*(4096/16) tiles.
// Inner loop: 2x b128 A loads + 32B B load + 1 WMMA (no VALU conversions).
// ---------------------------------------------------------------------------
__global__ __launch_bounds__(128)
void gemm_bias_kernel(const bf16_t* __restrict__ A, int lda,
                      const bf16_t* __restrict__ W,
                      const float* __restrict__ bias,
                      float* __restrict__ C) {
    const int wid  = (blockIdx.x * blockDim.x + threadIdx.x) >> 5;
    const int lane = threadIdx.x & 31;
    const int mTile = wid & 3;        // 0..3   (M = 64)
    const int nTile = wid >> 2;       // 0..255 (N = 4096)
    const int m0 = mTile * 16;
    const int n0 = nTile * 16;
    const int g  = lane >> 4;         // lane group (0,1)
    const int r  = lane & 15;

    typedef __attribute__((ext_vector_type(8))) __bf16 v8bf;
    const bf16_t* Arow = A + (size_t)(m0 + r) * (size_t)lda;  // A: M = lane%16
    const bf16_t* Wrow = W + (size_t)(n0 + r) * (size_t)HID;  // B: N = lane%16

    v8f acc = {};
    for (int k0 = 0; k0 < HID; k0 += 32) {
        // A fragment (contiguous K order, matches B):
        //   halves 0..7  : K = k0 + g*16 + 0..7     (contiguous 16B)
        //   halves 8..15 : K = k0 + g*16 + 8..15    (contiguous 16B)
        const v8bf a_lo = *(const v8bf*)(Arow + k0 + g * 16);
        const v8bf a_hi = *(const v8bf*)(Arow + k0 + g * 16 + 8);
        v16bf a;
#pragma unroll
        for (int i = 0; i < 8; ++i) { a[i] = a_lo[i]; a[8 + i] = a_hi[i]; }
        // B fragment: halves 0..15 : K = k0 + g*16 + 0..15 (contiguous 32B)
        const v16bf b = *(const v16bf*)(Wrow + k0 + g * 16);
        acc = __builtin_amdgcn_wmma_f32_16x16x32_bf16(
            /*neg_a=*/false, a, /*neg_b=*/false, b,
            /*c_mod=*/(short)0, acc, /*reuse_a=*/false, /*reuse_b=*/false);
    }

    // C/D layout: VGPR j -> row m0 + g*8 + j, col n0 + lane%16
    const int col = n0 + r;
    const float bv = bias[col];
#pragma unroll
    for (int j = 0; j < 8; ++j)
        C[(size_t)(m0 + g * 8 + j) * G4 + col] = acc[j] + bv;
}

// ---------------------------------------------------------------------------
// Fused cell: gates = LN4096(t1)*gx+bx + t2 ; LSTM update with LN(c), LN(h).
// One block per batch row; 256 threads, 4 hidden elements each.
// Also emits bf16 copy of new h (GEMM operand for next step / next layer).
// ---------------------------------------------------------------------------
__device__ __forceinline__ float sigmoidf_(float x) { return 1.0f / (1.0f + __expf(-x)); }

__global__ __launch_bounds__(256)
void cell_kernel(const float* __restrict__ t1, const float* __restrict__ t2,
                 const float* __restrict__ gx, const float* __restrict__ bx,
                 const float* __restrict__ gc, const float* __restrict__ bc,
                 const float* __restrict__ gh, const float* __restrict__ bh,
                 float* __restrict__ h, float* __restrict__ c,
                 bf16_t* __restrict__ hb,
                 float* __restrict__ out, long out_stride) {
    __shared__ float s1[256];
    __shared__ float s2[256];
    const int b   = blockIdx.x;
    const int tid = threadIdx.x;
    const float* r1 = t1 + (size_t)b * G4;
    const float* r2 = t2 + (size_t)b * G4;

    // ---- LN over 4096 of t1 ----
    float s = 0.f, q = 0.f;
    for (int i = tid; i < G4; i += 256) { float v = r1[i]; s += v; q += v * v; }
    s1[tid] = s; s2[tid] = q; __syncthreads();
    for (int off = 128; off > 0; off >>= 1) {
        if (tid < off) { s1[tid] += s1[tid + off]; s2[tid] += s2[tid + off]; }
        __syncthreads();
    }
    const float mean1 = s1[0] * (1.f / 4096.f);
    const float rstd1 = rsqrtf(s2[0] * (1.f / 4096.f) - mean1 * mean1 + 1e-5f);
    __syncthreads();

    // ---- gates + c update ----
    float cn[4], og[4];
    s = 0.f; q = 0.f;
#pragma unroll
    for (int qi = 0; qi < 4; ++qi) {
        const int j = tid + qi * 256;
        const float ig = (r1[j         ] - mean1) * rstd1 * gx[j         ] + bx[j         ] + r2[j         ];
        const float fg = (r1[j + 1*HID ] - mean1) * rstd1 * gx[j + 1*HID ] + bx[j + 1*HID ] + r2[j + 1*HID ];
        const float gg = (r1[j + 2*HID ] - mean1) * rstd1 * gx[j + 2*HID ] + bx[j + 2*HID ] + r2[j + 2*HID ];
        const float ov = (r1[j + 3*HID ] - mean1) * rstd1 * gx[j + 3*HID ] + bx[j + 3*HID ] + r2[j + 3*HID ];
        const float cv = sigmoidf_(fg) * c[(size_t)b * HID + j] + sigmoidf_(ig) * tanhf(gg);
        cn[qi] = cv; og[qi] = ov;
        s += cv; q += cv * cv;
    }
    s1[tid] = s; s2[tid] = q; __syncthreads();
    for (int off = 128; off > 0; off >>= 1) {
        if (tid < off) { s1[tid] += s1[tid + off]; s2[tid] += s2[tid + off]; }
        __syncthreads();
    }
    const float mean2 = s1[0] * (1.f / 1024.f);
    const float rstd2 = rsqrtf(s2[0] * (1.f / 1024.f) - mean2 * mean2 + 1e-5f);
    __syncthreads();

    // ---- LN(c), h pre-activation ----
    float hv[4];
    s = 0.f; q = 0.f;
#pragma unroll
    for (int qi = 0; qi < 4; ++qi) {
        const int j = tid + qi * 256;
        const float cl = (cn[qi] - mean2) * rstd2 * gc[j] + bc[j];
        c[(size_t)b * HID + j] = cl;
        const float hvv = sigmoidf_(og[qi]) * tanhf(cl);
        hv[qi] = hvv;
        s += hvv; q += hvv * hvv;
    }
    s1[tid] = s; s2[tid] = q; __syncthreads();
    for (int off = 128; off > 0; off >>= 1) {
        if (tid < off) { s1[tid] += s1[tid + off]; s2[tid] += s2[tid + off]; }
        __syncthreads();
    }
    const float mean3 = s1[0] * (1.f / 1024.f);
    const float rstd3 = rsqrtf(s2[0] * (1.f / 1024.f) - mean3 * mean3 + 1e-5f);

    // ---- LN(h), write state (fp32 + bf16) + output ----
#pragma unroll
    for (int qi = 0; qi < 4; ++qi) {
        const int j = tid + qi * 256;
        const float hn = (hv[qi] - mean3) * rstd3 * gh[j] + bh[j];
        h[(size_t)b * HID + j] = hn;
        hb[(size_t)b * HID + j] = (bf16_t)hn;
        if (out) out[(size_t)b * out_stride + j] = hn;
    }
}

// ---------------------------------------------------------------------------
// final states -> tail of d_out (order: h0, c0, h1, c1)
// ---------------------------------------------------------------------------
__global__ __launch_bounds__(256)
void copy_finals_kernel(const float* h0, const float* c0,
                        const float* h1, const float* c1, float* dst) {
    int i = blockIdx.x * blockDim.x + threadIdx.x;
    if (i < BATCH * HID) {
        dst[i]                 = h0[i];
        dst[BATCH*HID   + i]   = c0[i];
        dst[2*BATCH*HID + i]   = h1[i];
        dst[3*BATCH*HID + i]   = c1[i];
    }
}

// ---------------------------------------------------------------------------
extern "C" void kernel_launch(void* const* d_in, const int* in_sizes, int n_in,
                              void* d_out, int out_size, void* d_ws, size_t ws_size,
                              hipStream_t stream) {
    if (n_in < 21) return;
    const float* x = (const float*)d_in[0];

    // Disambiguate pytree flatten order: insertion (Wih,bih,Whh,bhh,gx,bx,gc,bc,gh,bh)
    // vs sorted keys (Whh,Wih,bc,bh,bhh,bih,bx,gc,gh,gx). in_sizes[2]: 4096 -> insertion.
    const bool sortedOrder = (in_sizes[2] > 100000);
    int iWih, iBih, iWhh, iBhh, iGx, iBx, iGc, iBc, iGh, iBh;
    if (!sortedOrder) { iWih=0; iBih=1; iWhh=2; iBhh=3; iGx=4; iBx=5; iGc=6; iBc=7; iGh=8; iBh=9; }
    else              { iWhh=0; iWih=1; iBc=2;  iBh=3;  iBhh=4; iBih=5; iBx=6; iGc=7; iGh=8; iGx=9; }

    const float *Wih[2], *Bih[2], *Whh[2], *Bhh[2], *Gx[2], *Bx[2], *Gc[2], *Bc[2], *Gh[2], *Bh[2];
    for (int l = 0; l < 2; ++l) {
        const int base = 1 + 10 * l;
        Wih[l] = (const float*)d_in[base + iWih];
        Bih[l] = (const float*)d_in[base + iBih];
        Whh[l] = (const float*)d_in[base + iWhh];
        Bhh[l] = (const float*)d_in[base + iBhh];
        Gx[l]  = (const float*)d_in[base + iGx];
        Bx[l]  = (const float*)d_in[base + iBx];
        Gc[l]  = (const float*)d_in[base + iGc];
        Bc[l]  = (const float*)d_in[base + iBc];
        Gh[l]  = (const float*)d_in[base + iGh];
        Bh[l]  = (const float*)d_in[base + iBh];
    }

    // workspace layout (~36 MB)
    char* ws = (char*)d_ws;
    const size_t WSZ = (size_t)G4 * HID * sizeof(bf16_t);   // 8 MB per matrix
    bf16_t* WihB[2] = { (bf16_t*)(ws + 0*WSZ),  (bf16_t*)(ws + 2*WSZ) };
    bf16_t* WhhB[2] = { (bf16_t*)(ws + 1*WSZ),  (bf16_t*)(ws + 3*WSZ) };
    char* p = ws + 4*WSZ;
    float* t1 = (float*)p;                           p += (size_t)BATCH * G4 * 4;
    float* t2 = (float*)p;                           p += (size_t)BATCH * G4 * 4;
    float* hS[2]; float* cS[2]; bf16_t* hB[2]; bf16_t* xB;
    hS[0] = (float*)p;  p += (size_t)BATCH * HID * 4;
    cS[0] = (float*)p;  p += (size_t)BATCH * HID * 4;
    hS[1] = (float*)p;  p += (size_t)BATCH * HID * 4;
    cS[1] = (float*)p;  p += (size_t)BATCH * HID * 4;
    hB[0] = (bf16_t*)p; p += (size_t)BATCH * HID * 2;
    hB[1] = (bf16_t*)p; p += (size_t)BATCH * HID * 2;
    xB    = (bf16_t*)p; p += (size_t)BATCH * HID * 2;

    // weight precision conversion + state init
    for (int l = 0; l < 2; ++l) {
        f32_to_bf16_kernel<<<2048, 256, 0, stream>>>(Wih[l], WihB[l], G4 * HID);
        f32_to_bf16_kernel<<<2048, 256, 0, stream>>>(Whh[l], WhhB[l], G4 * HID);
    }
    init_states_kernel<<<256, 256, 0, stream>>>(hS[0], cS[0], hS[1], cS[1], hB[0], hB[1]);

    // recurrence
    for (int t = 0; t < SEQT; ++t) {
        gather_x_bf16_kernel<<<BATCH, 256, 0, stream>>>(x + (size_t)t * HID, xB);
        for (int l = 0; l < 2; ++l) {
            const bf16_t* Ain = (l == 0) ? xB : hB[0];
            gemm_bias_kernel<<<256, 128, 0, stream>>>(Ain, HID, WihB[l], Bih[l], t1);
            gemm_bias_kernel<<<256, 128, 0, stream>>>(hB[l], HID, WhhB[l], Bhh[l], t2);
            float* outp = (l == 1) ? ((float*)d_out + (size_t)t * HID) : nullptr;
            cell_kernel<<<BATCH, 256, 0, stream>>>(t1, t2, Gx[l], Bx[l], Gc[l], Bc[l],
                                                   Gh[l], Bh[l], hS[l], cS[l], hB[l],
                                                   outp, (long)SEQT * HID);
        }
    }

    copy_finals_kernel<<<256, 256, 0, stream>>>(hS[0], cS[0], hS[1], cS[1],
                                                (float*)d_out + (size_t)BATCH * SEQT * HID);
}
